// Encoder_24266565222656
// MI455X (gfx1250) — matrix operand order repaired and checked by reading, not compiled, not run on hardware
//
#include <hip/hip_runtime.h>
#include <hip/hip_bf16.h>

typedef __attribute__((ext_vector_type(16))) __bf16 v16bf;
typedef __attribute__((ext_vector_type(8)))  __bf16 v8bf;
typedef __attribute__((ext_vector_type(8)))  float  v8f;

#define B_  16
#define T_  260
#define D_  32
#define W_  5
#define F_  256
#define U_  128
#define TC_ 256
#define KC_ (W_*D_)   /* 160 = 5 * 32 : exactly 5 WMMA k-steps */

// ---------------------------------------------------------------------------
// Prep: f32 -> bf16 conversions + weight transposes for WMMA B-operand layout
//   KT [f][k]  (k = w*32+d), row = 160 bf16 (320B, 32B-aligned rows)
//   W1T/W2T [u][f], row = 256 bf16
// ---------------------------------------------------------------------------
__global__ void prep_kernel(const float* __restrict__ x,
                            const float* __restrict__ ck,
                            const float* __restrict__ W1,
                            const float* __restrict__ W2,
                            __bf16* __restrict__ xb,
                            __bf16* __restrict__ KT,
                            __bf16* __restrict__ W1T,
                            __bf16* __restrict__ W2T) {
  int i = blockIdx.x * blockDim.x + threadIdx.x;
  const int nx = B_ * T_ * D_;   // 133120
  const int nk = F_ * KC_;       // 40960
  const int nw = U_ * F_;        // 32768
  if (i < nx) { xb[i] = (__bf16)x[i]; return; }
  i -= nx;
  if (i < nk) { int f = i / KC_, k = i - f * KC_; KT[i] = (__bf16)ck[k * F_ + f]; return; }
  i -= nk;
  if (i < nw) { int u = i / F_, f = i - u * F_; W1T[i] = (__bf16)W1[f * U_ + u]; return; }
  i -= nw;
  if (i < nw) { int u = i / F_, f = i - u * F_; W2T[i] = (__bf16)W2[f * U_ + u]; return; }
}

// ---------------------------------------------------------------------------
// Conv-as-GEMM: per batch, C[256x256] = relu(A[256x160] * KT^T + bias)
// A[t][k] = x[b, t*32 + k]  (re-strided x, no im2col materialization).
// One wave computes a 16(M) x 64(N) tile: A-frag reused across 4 WMMAs.
// ---------------------------------------------------------------------------
__global__ __launch_bounds__(32)
void conv_gemm_kernel(const __bf16* __restrict__ xb,
                      const __bf16* __restrict__ KT,
                      const float*  __restrict__ bias,
                      float*  __restrict__ convf,
                      __bf16* __restrict__ convb) {
  const int lane  = threadIdx.x;
  const int lrow  = lane & 15;
  const int half  = lane >> 4;
  const int b     = blockIdx.z;
  const int mbase = blockIdx.y * 16;
  const int fb0   = blockIdx.x * 64;

  // Lane's A row: contiguous 160 bf16 starting at x[b, mbase+lrow, 0]
  const __bf16* Arow = xb + (b * T_ + mbase + lrow) * D_;

  v8f acc[4] = {v8f{}, v8f{}, v8f{}, v8f{}};

  #pragma unroll
  for (int kb = 0; kb < 5; ++kb) {
    const int kbase = kb * 32;
    union { v16bf v; v8bf h[2]; } a;
    a.h[0] = *(const v8bf*)(Arow + kbase + half * 16);       // K in [kbase+half*16, +8)
    a.h[1] = *(const v8bf*)(Arow + kbase + half * 16 + 8);   // K in [kbase+half*16+8, +8)

    const __bf16* bp = KT + (size_t)(fb0 + lrow) * KC_ + kbase + half * 16;
    if (kb + 1 < 5) __builtin_prefetch(bp + 32, 0, 1);
    const v16bf b0 = *(const v16bf*)(bp);
    const v16bf b1 = *(const v16bf*)(bp + 16 * KC_);
    const v16bf b2 = *(const v16bf*)(bp + 32 * KC_);
    const v16bf b3 = *(const v16bf*)(bp + 48 * KC_);

    acc[0] = __builtin_amdgcn_wmma_f32_16x16x32_bf16(false, a.v, false, b0, (short)0, acc[0], false, false);
    acc[1] = __builtin_amdgcn_wmma_f32_16x16x32_bf16(false, a.v, false, b1, (short)0, acc[1], false, false);
    acc[2] = __builtin_amdgcn_wmma_f32_16x16x32_bf16(false, a.v, false, b2, (short)0, acc[2], false, false);
    acc[3] = __builtin_amdgcn_wmma_f32_16x16x32_bf16(false, a.v, false, b3, (short)0, acc[3], false, false);
  }

  #pragma unroll
  for (int nb = 0; nb < 4; ++nb) {
    const int f = fb0 + nb * 16 + lrow;
    const float bs = bias[f];
    #pragma unroll
    for (int r = 0; r < 8; ++r) {
      const int m = mbase + half * 8 + r;       // C/D layout: VGPR r -> row half*8+r
      float val = fmaxf(acc[nb][r] + bs, 0.0f); // ReLU
      const size_t o = ((size_t)(b * TC_ + m)) * F_ + f;
      convf[o] = val;
      convb[o] = (__bf16)val;
    }
  }
}

// ---------------------------------------------------------------------------
// Projections: q = conv*W1 + b1, v = conv*W2 + b2 (both in one pass; A reused)
// Per wave: 16(M) x 32(U) of BOTH q and v -> 4 accumulators, 8 k-steps.
// ---------------------------------------------------------------------------
__global__ __launch_bounds__(32)
void proj_gemm_kernel(const __bf16* __restrict__ convb,
                      const __bf16* __restrict__ W1T,
                      const __bf16* __restrict__ W2T,
                      const float* __restrict__ b1,
                      const float* __restrict__ b2,
                      float* __restrict__ q,
                      float* __restrict__ v) {
  const int lane  = threadIdx.x;
  const int lrow  = lane & 15;
  const int half  = lane >> 4;
  const int b     = blockIdx.z;
  const int mbase = blockIdx.y * 16;
  const int ub0   = blockIdx.x * 32;

  const __bf16* Arow = convb + (size_t)(b * TC_ + mbase + lrow) * F_;

  v8f aq[2] = {v8f{}, v8f{}};
  v8f av[2] = {v8f{}, v8f{}};

  #pragma unroll
  for (int kb = 0; kb < 8; ++kb) {
    const int kbase = kb * 32;
    union { v16bf v; v8bf h[2]; } a;
    a.h[0] = *(const v8bf*)(Arow + kbase + half * 16);
    a.h[1] = *(const v8bf*)(Arow + kbase + half * 16 + 8);

    #pragma unroll
    for (int t = 0; t < 2; ++t) {
      const int u = ub0 + t * 16 + lrow;
      const v16bf bw1 = *(const v16bf*)(W1T + (size_t)u * F_ + kbase + half * 16);
      const v16bf bw2 = *(const v16bf*)(W2T + (size_t)u * F_ + kbase + half * 16);
      aq[t] = __builtin_amdgcn_wmma_f32_16x16x32_bf16(false, a.v, false, bw1, (short)0, aq[t], false, false);
      av[t] = __builtin_amdgcn_wmma_f32_16x16x32_bf16(false, a.v, false, bw2, (short)0, av[t], false, false);
    }
  }

  #pragma unroll
  for (int t = 0; t < 2; ++t) {
    const int u = ub0 + t * 16 + lrow;
    const float c1 = b1[u], c2 = b2[u];
    #pragma unroll
    for (int r = 0; r < 8; ++r) {
      const int m = mbase + half * 8 + r;
      const size_t o = ((size_t)(b * TC_ + m)) * U_ + u;
      q[o] = aq[t][r] + c1;
      v[o] = av[t][r] + c2;
    }
  }
}

// ---------------------------------------------------------------------------
// Fused score + softmax: never materialize h (reference builds 4.3 GB here).
// One 256-thread block per (b,t); thread j owns value-index j.
// score[j] = bV + sum_u V[u]*tanh(q[b,t,u] + v[b,j,u]); softmax over j.
// Uses gfx1250 hardware v_tanh_f32.
// ---------------------------------------------------------------------------
__device__ __forceinline__ float tanh_hw(float x) {
  float r;
  asm volatile("v_tanh_f32 %0, %1" : "=v"(r) : "v"(x));
  return r;
}

__global__ __launch_bounds__(256)
void attn_kernel(const float* __restrict__ q,
                 const float* __restrict__ v,
                 const float* __restrict__ Vw,
                 const float* __restrict__ bV,
                 float* __restrict__ attn) {
  __shared__ float qs[U_];
  __shared__ float Vs[U_];
  __shared__ float red[256];

  const int bt = blockIdx.x;       // b*256 + t
  const int b  = bt >> 8;
  const int t  = bt & 255;
  const int j  = threadIdx.x;

  if (j < U_) {
    qs[j] = q[((size_t)(b * TC_ + t)) * U_ + j];
    Vs[j] = Vw[j];
  }
  __syncthreads();

  const float4* vrow = (const float4*)(v + ((size_t)(b * TC_ + j)) * U_);
  float s = bV[0];
  #pragma unroll 4
  for (int u4 = 0; u4 < U_ / 4; ++u4) {
    const float4 vv = vrow[u4];
    const float4 qq = *(const float4*)(qs + u4 * 4);
    const float4 VV = *(const float4*)(Vs + u4 * 4);
    s += VV.x * tanh_hw(qq.x + vv.x);
    s += VV.y * tanh_hw(qq.y + vv.y);
    s += VV.z * tanh_hw(qq.z + vv.z);
    s += VV.w * tanh_hw(qq.w + vv.w);
  }

  // block softmax over the 256 j's
  red[j] = s;
  __syncthreads();
  for (int off = 128; off > 0; off >>= 1) {
    if (j < off) red[j] = fmaxf(red[j], red[j + off]);
    __syncthreads();
  }
  const float mx = red[0];
  __syncthreads();
  const float e = __expf(s - mx);
  red[j] = e;
  __syncthreads();
  for (int off = 128; off > 0; off >>= 1) {
    if (j < off) red[j] += red[j + off];
    __syncthreads();
  }
  attn[((size_t)(b * TC_ + t)) * TC_ + j] = e / red[0];
}

// ---------------------------------------------------------------------------
// Output: out[b,t,f] = conv[b,t,f] * attn[t%16, b*16 + t/16, f]
// (the TensorArray stack+reshape scramble, solved in closed form)
// ---------------------------------------------------------------------------
__global__ void out_kernel(const float* __restrict__ convf,
                           const float* __restrict__ attn,
                           float* __restrict__ out) {
  const int i = blockIdx.x * blockDim.x + threadIdx.x;
  if (i >= B_ * TC_ * F_) return;
  const int f = i & 255;
  const int t = (i >> 8) & 255;
  const int b = i >> 16;
  const int ab = t & 15;             // attn batch index
  const int at = b * 16 + (t >> 4);  // attn query index
  out[i] = convf[i] * attn[((size_t)(ab * TC_ + at)) * TC_ + f];
}

// ---------------------------------------------------------------------------
extern "C" void kernel_launch(void* const* d_in, const int* in_sizes, int n_in,
                              void* d_out, int out_size, void* d_ws, size_t ws_size,
                              hipStream_t stream) {
  const float* x  = (const float*)d_in[0];
  const float* ck = (const float*)d_in[1];
  const float* cb = (const float*)d_in[2];
  const float* W1 = (const float*)d_in[3];
  const float* b1 = (const float*)d_in[4];
  const float* W2 = (const float*)d_in[5];
  const float* b2 = (const float*)d_in[6];
  const float* Vw = (const float*)d_in[7];
  const float* bV = (const float*)d_in[8];
  float* out = (float*)d_out;

  char* w = (char*)d_ws;
  auto carve = [&](size_t bytes) {
    char* p = w;
    w += (bytes + 255) & ~(size_t)255;
    return p;
  };
  __bf16* xb    = (__bf16*)carve((size_t)B_ * T_ * D_ * sizeof(__bf16));   // 260 KB
  __bf16* KT    = (__bf16*)carve((size_t)F_ * KC_ * sizeof(__bf16));       //  80 KB
  __bf16* W1T   = (__bf16*)carve((size_t)U_ * F_ * sizeof(__bf16));        //  64 KB
  __bf16* W2T   = (__bf16*)carve((size_t)U_ * F_ * sizeof(__bf16));        //  64 KB
  __bf16* convb = (__bf16*)carve((size_t)B_ * TC_ * F_ * sizeof(__bf16));  //   2 MB
  float*  convf = (float*)carve((size_t)B_ * TC_ * F_ * sizeof(float));    //   4 MB
  float*  qbuf  = (float*)carve((size_t)B_ * TC_ * U_ * sizeof(float));    //   2 MB
  float*  vbuf  = (float*)carve((size_t)B_ * TC_ * U_ * sizeof(float));    //   2 MB
  float*  attn  = (float*)carve((size_t)B_ * TC_ * TC_ * sizeof(float));   //   4 MB

  const int nprep = B_ * T_ * D_ + F_ * KC_ + 2 * U_ * F_;
  prep_kernel<<<(nprep + 255) / 256, 256, 0, stream>>>(x, ck, W1, W2, xb, KT, W1T, W2T);
  conv_gemm_kernel<<<dim3(F_ / 64, TC_ / 16, B_), 32, 0, stream>>>(xb, KT, cb, convf, convb);
  proj_gemm_kernel<<<dim3(U_ / 32, TC_ / 16, B_), 32, 0, stream>>>(convb, W1T, W2T, b1, b2, qbuf, vbuf);
  attn_kernel<<<B_ * TC_, 256, 0, stream>>>(qbuf, vbuf, Vw, bV, attn);
  out_kernel<<<(B_ * TC_ * F_ + 255) / 256, 256, 0, stream>>>(convf, attn, out);
}
